// LBLHighwayBiLm_10419590660605
// MI455X (gfx1250) — matrix operand order repaired and checked by reading, not compiled, not run on hardware
//
#include <hip/hip_runtime.h>

// ---------------------------------------------------------------------------
// LBLHighwayBiLm for MI455X (gfx1250, wave32, WMMA + async global->LDS).
// M = B*S = 16384, K = H = 1024, N = 2H = 2048. 8 highway GEMMs (~550 GFLOP)
// dominate -> bf16 WMMA w/ f32 accumulate, fused highway epilogue.
// Weights pre-transposed to [N][K] bf16 so both A and B tiles stage with b128.
// GEMM: block tile 128(M) x 32(N of nl) + 32(N of g), double-buffered LDS fed
// by GLOBAL_LOAD_ASYNC_TO_LDS_B128 (ASYNCcnt pipeline) when available.
// Workspace: [ Wt bf16 32MB | x0 bf16 32MB | x1 bf16 32MB ] = 96MB.
// ---------------------------------------------------------------------------

typedef __attribute__((ext_vector_type(16))) __bf16 v16bf;
typedef __attribute__((ext_vector_type(8)))  float  v8f;

static constexpr int    HDIM  = 1024;
static constexpr int    TWOH  = 2048;
static constexpr int    SEQ   = 512;
static constexpr int    MROWS = 16384;     // B*S
static constexpr int    KDIM  = 1024;      // == HDIM
static constexpr int    WWIN  = 4;         // WIDTH
static constexpr size_t MATSZ = (size_t)HDIM * TWOH;   // elems per highway matrix

#if __has_builtin(__builtin_amdgcn_global_load_async_to_lds_b128) && \
    __has_builtin(__builtin_amdgcn_s_wait_asynccnt)
#define USE_ASYNC_LDS 1
// Param 0 is 'int __vector(4) __device__*' per clang diagnostic -> int4 in AS1.
typedef int v4i_t __attribute__((vector_size(16)));
typedef __attribute__((address_space(1))) v4i_t gv4i;
typedef __attribute__((address_space(3))) v4i_t lv4i;
#define ASYNC_B128(gp, lp) \
    __builtin_amdgcn_global_load_async_to_lds_b128((gv4i*)(gp), (lv4i*)(lp), 0, 0)
#else
#define USE_ASYNC_LDS 0
#endif

// ---- weight f32 -> bf16 + transpose: W[l,i][k][n] -> Wt[mat][n][k] ---------
// mat = dir*4 + l*2 + i.  Tiled 32x32 via LDS, coalesced loads and stores.
__global__ __launch_bounds__(256) void transpose_w_kernel(
    const float* __restrict__ fw, const float* __restrict__ bw,
    __bf16* __restrict__ wt)
{
    const int mat = blockIdx.z;                       // 0..7
    const float* src = (mat >= 4 ? bw : fw) + (size_t)(mat & 3) * MATSZ;
    __bf16* dst = wt + (size_t)mat * MATSZ;           // [n][k]
    const int k0 = blockIdx.x * 32;
    const int n0 = blockIdx.y * 32;

    __shared__ __bf16 t[32][33];
    const int row = threadIdx.x >> 5;                 // 0..7
    const int col = threadIdx.x & 31;                 // 0..31
#pragma unroll
    for (int r = 0; r < 4; ++r) {
        const int rr = row + r * 8;
        t[rr][col] = (__bf16)src[(size_t)(k0 + rr) * TWOH + n0 + col];
    }
    __syncthreads();
#pragma unroll
    for (int r = 0; r < 4; ++r) {
        const int rr = row + r * 8;
        dst[(size_t)(n0 + rr) * KDIM + k0 + col] = t[col][rr];
    }
}

// ---- 5-tap window sum over padded sequence, emit bf16 activation matrix ----
// padded = [fw_pad(l) | src | bw_pad(l)] for BOTH directions (reference quirk).
__global__ __launch_bounds__(256) void window_kernel(
    const float* __restrict__ src, int sstride,
    const float* __restrict__ padF, const float* __restrict__ padB,
    const float* __restrict__ w5, int off, __bf16* __restrict__ xout)
{
    size_t idx = (size_t)blockIdx.x * 256 + threadIdx.x;     // 0 .. 16777215
    int h = (int)(idx & (HDIM - 1));
    int m = (int)(idx >> 10);            // b*S + s
    int s = m & (SEQ - 1);
    float acc = 0.f;
#pragma unroll
    for (int k = 0; k <= WWIN; ++k) {
        int j = off + s + k;             // index into padded sequence
        float v;
        if (j < WWIN)            v = padF[j * HDIM + h];
        else if (j < WWIN + SEQ) v = src[(size_t)(m - s + (j - WWIN)) * sstride + h];
        else                     v = padB[(j - WWIN - SEQ) * HDIM + h];
        acc += w5[k] * v;
    }
    xout[idx] = (__bf16)acc;
}

// ---- fused highway GEMM: proj = X @ W + b ; x' = sig(g)*x + (1-sig(g))*relu(nl)
// Block 256 thr = 8 waves. Tile: M=128, nl cols [n0,n0+32), g cols [H+n0,+32).
// Wave (wr 0..3, wc 0..1): rows wr*32+{0,16}, cols wc*16; 4 accumulators.
__global__ __launch_bounds__(256) void hw_gemm_kernel(
    const __bf16* __restrict__ Xbf,    // [16384][1024] bf16
    const __bf16* __restrict__ Wt,     // [2048][1024]  bf16 (transposed)
    const float*  __restrict__ bias,   // [2048]
    __bf16* __restrict__ xout_bf,      // [16384][1024] or null
    float*  __restrict__ out_f32,      // strided or null
    long out_stride,
    const float* __restrict__ cache,   // residual, strided or null
    long cache_stride)
{
    __shared__ __align__(16) __bf16 As[2][128][40];   // [buf][m][k], +8 pad
    __shared__ __align__(16) __bf16 Bt[2][64][40];    // [buf][n' 0..31 nl,32..63 g][k]

    const int tid  = threadIdx.x;
    const int lane = tid & 31;
    const int wave = tid >> 5;
    const int wr   = wave >> 1;            // 0..3 : M sub-tile (32 rows)
    const int wc   = wave & 1;             // 0..1 : N sub-tile (16 cols)
    const bool hi  = lane >= 16;
    const int lm   = lane & 15;

    const int m0 = blockIdx.x * 128;
    const int n0 = blockIdx.y * 32;

    // staging indices: A = 128x32 (8KB, 2 b128/thr), B = 64x32 (4KB, 1 b128/thr)
    const int arow = tid >> 2;               // 0..63 (and +64)
    const int acol = (tid & 3) * 8;          // 0,8,16,24
    const int brow = tid >> 2;               // n' 0..63
    const int bcol = (tid & 3) * 8;
    const int bgrow = n0 + (brow & 31) + ((brow >> 5) * HDIM);   // Wt row

    const __bf16* gA = Xbf + (size_t)(m0 + arow) * KDIM + acol;
    const __bf16* gB = Wt + (size_t)bgrow * KDIM + bcol;

    v8f cnl[2] = {v8f{}, v8f{}};
    v8f cg [2] = {v8f{}, v8f{}};

    auto stage = [&](int p, int k0) {
#if USE_ASYNC_LDS
        ASYNC_B128(gA + k0,                    &As[p][arow][acol]);
        ASYNC_B128(gA + k0 + (size_t)64*KDIM,  &As[p][arow + 64][acol]);
        ASYNC_B128(gB + k0,                    &Bt[p][brow][bcol]);
#else
        const uint4 a0 = *(const uint4*)(gA + k0);
        const uint4 a1 = *(const uint4*)(gA + k0 + (size_t)64 * KDIM);
        const uint4 b0 = *(const uint4*)(gB + k0);
        *(uint4*)(&As[p][arow][acol])      = a0;
        *(uint4*)(&As[p][arow + 64][acol]) = a1;
        *(uint4*)(&Bt[p][brow][bcol])      = b0;
#endif
    };

    stage(0, 0);
    int p = 0;
    for (int k0 = 0; k0 < KDIM; k0 += 32) {
        if (k0 + 32 < KDIM) {
            stage(p ^ 1, k0 + 32);           // overlap next copy with compute
#if USE_ASYNC_LDS
            __builtin_amdgcn_s_wait_asynccnt(3);   // buf p complete (in-order)
#endif
        } else {
#if USE_ASYNC_LDS
            __builtin_amdgcn_s_wait_asynccnt(0);
#endif
        }
        __syncthreads();

        // A fragments (contiguous k order, same as B):
        // lanes 0-15: K 0..15, lanes 16-31: K 16..31
        v16bf a0, a1;
        {
            const int ar = wr * 32 + lm;
            const int kb = hi ? 16 : 0;
#pragma unroll
            for (int hh = 0; hh < 8; ++hh)  { a0[hh] = As[p][ar][kb + hh];
                                              a1[hh] = As[p][ar + 16][kb + hh]; }
#pragma unroll
            for (int hh = 8; hh < 16; ++hh) { a0[hh] = As[p][ar][kb + hh];
                                              a1[hh] = As[p][ar + 16][kb + hh]; }
        }
        // B fragments (lanes 0-15: K 0..15, lanes 16-31: K 16..31)
        v16bf bnl, bg;
        {
            const int kb    = hi ? 16 : 0;
            const int brow0 = wc * 16 + lm;
#pragma unroll
            for (int hh = 0; hh < 16; ++hh) bnl[hh] = Bt[p][brow0][kb + hh];
#pragma unroll
            for (int hh = 0; hh < 16; ++hh) bg[hh]  = Bt[p][32 + brow0][kb + hh];
        }
        cnl[0] = __builtin_amdgcn_wmma_f32_16x16x32_bf16(false, a0, false, bnl, (short)0, cnl[0], false, false);
        cnl[1] = __builtin_amdgcn_wmma_f32_16x16x32_bf16(false, a1, false, bnl, (short)0, cnl[1], false, false);
        cg[0]  = __builtin_amdgcn_wmma_f32_16x16x32_bf16(false, a0, false, bg,  (short)0, cg[0],  false, false);
        cg[1]  = __builtin_amdgcn_wmma_f32_16x16x32_bf16(false, a1, false, bg,  (short)0, cg[1],  false, false);
        __syncthreads();                     // frags consumed before buf reuse
        p ^= 1;
    }

    // ---- fused highway epilogue ----
    // C/D layout: VGPR r -> row M = r (+8 for lanes 16-31), col N = lane%16
    const int n = n0 + wc * 16 + lm;                 // col in [0,H)
    const float bnl_b = bias[n];
    const float bg_b  = bias[HDIM + n];
#pragma unroll
    for (int sub = 0; sub < 2; ++sub) {
#pragma unroll
        for (int r = 0; r < 8; ++r) {
            const int m = m0 + wr * 32 + sub * 16 + r + (hi ? 8 : 0);
            const float nlv = fmaxf(cnl[sub][r] + bnl_b, 0.f);
            float gv = cg[sub][r] + bg_b;
            gv = 1.0f / (1.0f + __expf(-gv));
            const float xv = (float)Xbf[(size_t)m * KDIM + n];
            float o = gv * xv + (1.f - gv) * nlv;
            if (cache)   o += cache[(size_t)m * cache_stride + n];
            if (out_f32) out_f32[(size_t)m * out_stride + n] = o;
            if (xout_bf) xout_bf[(size_t)m * KDIM + n] = (__bf16)o;
        }
    }
}

// ---------------------------------------------------------------------------
extern "C" void kernel_launch(void* const* d_in, const int* in_sizes, int n_in,
                              void* d_out, int out_size, void* d_ws, size_t ws_size,
                              hipStream_t stream)
{
    const float* inputs  = (const float*)d_in[0];
    // d_in[1] = masks (all ones, unused by reference math)
    const float* fw_pad  = (const float*)d_in[2];
    const float* bw_pad  = (const float*)d_in[3];
    const float* fw_w    = (const float*)d_in[4];
    const float* bw_w    = (const float*)d_in[5];
    const float* fw_hw_W = (const float*)d_in[6];
    const float* fw_hw_b = (const float*)d_in[7];
    const float* bw_hw_W = (const float*)d_in[8];
    const float* bw_hw_b = (const float*)d_in[9];
    float* out = (float*)d_out;                      // [L][B][S][2H]

    __bf16* Wt = (__bf16*)d_ws;                                   // 8*MATSZ bf16
    __bf16* x0 = (__bf16*)((char*)d_ws + (size_t)8 * MATSZ * 2);  // 32MB in
    __bf16* x1 = x0 + (size_t)MROWS * KDIM;

    transpose_w_kernel<<<dim3(KDIM / 32, TWOH / 32, 8), 256, 0, stream>>>(
        fw_hw_W, bw_hw_W, Wt);

    const dim3 ggrid(MROWS / 128, HDIM / 32);        // (128, 32)
    for (int l = 0; l < 2; ++l) {
        for (int dir = 0; dir < 2; ++dir) {
            const float* w5 = (dir ? bw_w : fw_w) + l * 5;
            const float* src; int sstride;
            if (l == 0) { src = inputs; sstride = HDIM; }
            else        { src = out + (size_t)(l - 1) * MROWS * TWOH + dir * HDIM; sstride = TWOH; }

            window_kernel<<<65536, 256, 0, stream>>>(
                src, sstride,
                fw_pad + (size_t)l * WWIN * HDIM,
                bw_pad + (size_t)l * WWIN * HDIM,
                w5, dir ? WWIN : 0, x0);

            const __bf16* W0 = Wt + (size_t)(dir * 4 + l * 2 + 0) * MATSZ;
            const __bf16* W1 = Wt + (size_t)(dir * 4 + l * 2 + 1) * MATSZ;
            const float* bvec = dir ? bw_hw_b : fw_hw_b;
            const float* b0 = bvec + (size_t)(l * 2 + 0) * TWOH;
            const float* b1 = bvec + (size_t)(l * 2 + 1) * TWOH;

            // highway 0: x0 -> x1 (bf16)
            hw_gemm_kernel<<<ggrid, 256, 0, stream>>>(
                x0, W0, b0, x1, nullptr, 0, nullptr, 0);

            // highway 1: x1 -> d_out[l] (f32), fused residual from d_out[l-1]
            float* o = out + (size_t)l * MROWS * TWOH + dir * HDIM;
            const float* cache = l ? (out + (size_t)(l - 1) * MROWS * TWOH + dir * HDIM)
                                   : nullptr;
            hw_gemm_kernel<<<ggrid, 256, 0, stream>>>(
                x1, W1, b1, nullptr, o, TWOH, cache, TWOH);
        }
    }
}
